// Self_Attention_3023656977021
// MI455X (gfx1250) — matrix operand order repaired and checked by reading, not compiled, not run on hardware
//
#include <hip/hip_runtime.h>

// MI455X / gfx1250, wave32. Fused SAGAN self-attention.
// d_out = [ out (4*64*64*64 f32) | attention (4*4096*4096 f32) ]
// Workspace (needs ~4.2 MB): Qp/Kp (bf16 B*N*32), Vp (bf16 B*64*4096).
//
// Roofline: ~272 MB of mandatory output writes vs ~17 GFLOP (incl. energy
// recompute) -> bandwidth bound on the attention write. Design: write the
// attention matrix exactly once with b128 stores, everything else fused,
// matmuls on v_wmma_f32_16x16x32_bf16 (fp32 accumulate).

typedef __attribute__((ext_vector_type(16))) __bf16 v16bf;
typedef __attribute__((ext_vector_type(8)))  __bf16 v8bf;
typedef __attribute__((ext_vector_type(8)))  float  v8f;

#define WMMA_BF16(A, B, C) \
  __builtin_amdgcn_wmma_f32_16x16x32_bf16(false, (A), false, (B), (short)0, (C), false, false)

static __device__ __forceinline__ v16bf cat8(v8bf lo, v8bf hi) {
  return __builtin_shufflevector(lo, hi, 0,1,2,3,4,5,6,7,8,9,10,11,12,13,14,15);
}

namespace {
constexpr int NB = 4;        // batch
constexpr int NN = 4096;     // N = W*H
}

// ---------------------------------------------------------------------------
// Kernel 1: q/k/v projections via WMMA, scatter into WMMA-native layouts.
//   Qp[b][n][d]  (n-major rows of 32, bf16)  -- A-operand layout for energy
//   Kp[b][m][d]  (m-major rows of 32, bf16)  -- gathered as B-operand
//   Vp[b][c][m]  (c-major rows of 4096,bf16) -- A-operand for out-matmul
// ---------------------------------------------------------------------------
__global__ __launch_bounds__(128) void proj_kernel(
    const float* __restrict__ x,
    const float* __restrict__ Wq, const float* __restrict__ bq,
    const float* __restrict__ Wk, const float* __restrict__ bk,
    const float* __restrict__ Wv, const float* __restrict__ bv,
    __bf16* __restrict__ Qp, __bf16* __restrict__ Kp, __bf16* __restrict__ Vp)
{
  const int lane = threadIdx.x & 31;
  const int wid  = threadIdx.x >> 5;
  const int t    = blockIdx.x * 4 + wid;     // 1024 pixel tiles
  const int b    = t >> 8;
  const int p0   = (t & 255) * 16;           // pixel base within batch

  const int l  = lane & 15;
  const int hs = lane >> 4;

  // A operands: x tile (16 px x 64 ch) -> two K=32 chunks (A layout:
  // lo-lanes hold K {0..7,16..23}, hi-lanes {8..15,24..31}).
  const float* xr = x + ((size_t)(b * NN + p0 + l) << 6);
  v16bf a0, a1;
  #pragma unroll
  for (int e = 0; e < 8; ++e) {
    a0[e]     = (__bf16)xr[hs * 8 + e];
    a0[8 + e] = (__bf16)xr[16 + hs * 8 + e];
    a1[e]     = (__bf16)xr[32 + hs * 8 + e];
    a1[8 + e] = (__bf16)xr[48 + hs * 8 + e];
  }

  // Weight B-operand gather (B layout: lane = column, lo-lanes K {0..7,16..23},
  // hi-lanes K {8..15,24..31} relative to k0).
  auto loadWB = [&](const float* W, int ld, int col0, int k0) -> v16bf {
    v16bf r;
    const float* wp = W + (size_t)(k0 + hs * 8) * ld + (col0 + l);
    #pragma unroll
    for (int e = 0; e < 16; ++e) r[e] = (__bf16)wp[(e + (e & 8)) * ld];
    return r;
  };
  auto tileOut = [&](const float* W, int ld, int col0) -> v8f {
    v8f acc = {};
    acc = WMMA_BF16(a0, loadWB(W, ld, col0, 0),  acc);
    acc = WMMA_BF16(a1, loadWB(W, ld, col0, 32), acc);
    return acc;
  };

  // q/k scatter: pq[b,n,d] = q[b, 2d+(n>>11), (n>>5)&63, n&31]
  auto storeQK = [&](v8f acc, const float* bias, int col0, __bf16* dst) {
    const int cq = col0 + l;
    const float bb = bias[cq];
    #pragma unroll
    for (int j = 0; j < 8; ++j) {
      const int p = p0 + hs * 8 + j;           // D row
      const int w = p >> 6, h = p & 63;
      const int n = ((w & 1) << 11) + (h << 5) + cq;
      const int d = w >> 1;
      dst[((size_t)(b * NN + n) << 5) + d] = (__bf16)(acc[j] + bb);
    }
  };

  #pragma unroll
  for (int ct = 0; ct < 2; ++ct) storeQK(tileOut(Wq, 32, ct * 16), bq, ct * 16, Qp);
  #pragma unroll
  for (int ct = 0; ct < 2; ++ct) storeQK(tileOut(Wk, 32, ct * 16), bk, ct * 16, Kp);

  // v: pv[b,c,m] = v[b, c=w, m>>6, m&63]  (clean mapping)
  #pragma unroll
  for (int ct = 0; ct < 4; ++ct) {
    v8f acc = tileOut(Wv, 64, ct * 16);
    const int cv = ct * 16 + l;
    const float bb = bv[cv];
    #pragma unroll
    for (int j = 0; j < 8; ++j) {
      const int p = p0 + hs * 8 + j;
      const int w = p >> 6, h = p & 63;
      Vp[((size_t)(b * 64 + w) << 12) + (h << 6) + cv] = (__bf16)(acc[j] + bb);
    }
  }
}

// ---------------------------------------------------------------------------
// Kernel 2: fused energy -> softmax -> attention write -> out accumulation.
// One wave owns one (b, 16-row n-tile). Two passes over m; energy recompute
// is ~free vs the 268 MB attention write, which happens exactly once and
// with b128 stores via an fp32 LDS transpose.
// ---------------------------------------------------------------------------
__global__ __launch_bounds__(128) void attn_kernel(
    const __bf16* __restrict__ Qp, const __bf16* __restrict__ Kp,
    const __bf16* __restrict__ Vp,
    const float* __restrict__ x, const float* __restrict__ gamma,
    float* __restrict__ out, float* __restrict__ attn)
{
  // wave-private 16x32 fp32 P staging; row stride 36 floats (144 B, 16B-aligned
  // chunks, banks well spread).
  __shared__ float ldsP[4][16 * 36];

  const int lane = threadIdx.x & 31;
  const int wid  = threadIdx.x >> 5;
  const int t    = blockIdx.x * 4 + wid;   // 1024 n-tiles
  const int b    = t >> 8;
  const int n0   = (t & 255) * 16;
  const int l    = lane & 15;
  const int hs   = lane >> 4;

  // Q A-operand, kept in registers for both passes.
  const __bf16* qr = Qp + ((size_t)(b * NN + n0 + l) << 5);
  const v16bf qa = cat8(*(const v8bf*)(qr + hs * 16), *(const v8bf*)(qr + hs * 16 + 8));

  const __bf16* kbase = Kp + ((size_t)(b * NN + l) << 5) + hs * 16;

  // ---- Pass 1: online softmax stats (per-lane, then 16-lane merge) ----
  float m8[8], s8[8];
  #pragma unroll
  for (int j = 0; j < 8; ++j) { m8[j] = -3.0e38f; s8[j] = 0.0f; }

  for (int m0 = 0; m0 < NN; m0 += 16) {
    const __bf16* kr = kbase + ((size_t)m0 << 5);
    __builtin_prefetch(kr + (16 << 5), 0, 1);           // next K chunk
    const v16bf kb = cat8(*(const v8bf*)kr, *(const v8bf*)(kr + 8));
    v8f e = {};
    e = WMMA_BF16(qa, kb, e);
    #pragma unroll
    for (int j = 0; j < 8; ++j) {
      const float mn = fmaxf(m8[j], e[j]);
      s8[j] = s8[j] * __expf(m8[j] - mn) + __expf(e[j] - mn);
      m8[j] = mn;
    }
  }
  #pragma unroll
  for (int mask = 1; mask <= 8; mask <<= 1) {
    #pragma unroll
    for (int j = 0; j < 8; ++j) {
      const float mo = __shfl_xor(m8[j], mask, 32);
      const float so = __shfl_xor(s8[j], mask, 32);
      const float mn = fmaxf(m8[j], mo);
      s8[j] = s8[j] * __expf(m8[j] - mn) + so * __expf(mo - mn);
      m8[j] = mn;
    }
  }
  float rs8[8];
  #pragma unroll
  for (int j = 0; j < 8; ++j) rs8[j] = 1.0f / s8[j];

  // ---- Pass 2 ----
  v8f acc[4] = {v8f{}, v8f{}, v8f{}, v8f{}};
  float* arow = attn + ((size_t)b << 24) + ((size_t)n0 << 12);
  float* lp = &ldsP[wid][0];

  const int ar  = lane >> 1;          // attention-store row owned by this lane
  const int ac0 = (lane & 1) * 16;    // ...and its 16-column chunk

  for (int m0 = 0; m0 < NN; m0 += 32) {
    // E tiles -> normalized P staged in LDS (fp32)
    #pragma unroll
    for (int sub = 0; sub < 2; ++sub) {
      const int mm = m0 + sub * 16;
      const __bf16* kr = kbase + ((size_t)mm << 5);
      const v16bf kb = cat8(*(const v8bf*)kr, *(const v8bf*)(kr + 8));
      v8f e = {};
      e = WMMA_BF16(qa, kb, e);
      #pragma unroll
      for (int j = 0; j < 8; ++j) {
        const int r = hs * 8 + j;     // tile row (n)
        lp[r * 36 + sub * 16 + l] = __expf(e[j] - m8[j]) * rs8[j];
      }
    }
    asm volatile("s_wait_dscnt 0" ::: "memory");   // stores visible to wave

    // Attention write: each lane owns 16 consecutive floats of one row ->
    // 4x global_store_b128 (128 B contiguous per lane pair).
    {
      float* dst = arow + (size_t)ar * 4096 + m0 + ac0;
      const float* src = lp + ar * 36 + ac0;
      #pragma unroll
      for (int q4 = 0; q4 < 4; ++q4)
        *(float4*)(dst + q4 * 4) = *(const float4*)(src + q4 * 4);
    }

    // B operand P^T (32m x 16n): lane = n column (row of P), lo-lanes hold
    // m 0..15, hi-lanes m 16..31 -> 16 consecutive floats, cvt to bf16.
    v16bf pb;
    {
      const float* src = lp + l * 36 + hs * 16;
      float f[16];
      #pragma unroll
      for (int q4 = 0; q4 < 4; ++q4)
        *(float4*)(f + q4 * 4) = *(const float4*)(src + q4 * 4);
      #pragma unroll
      for (int e = 0; e < 16; ++e) pb[e] = (__bf16)f[e];
    }

    #pragma unroll
    for (int ct = 0; ct < 4; ++ct) {               // 4 c-tiles of 16
      const __bf16* vr = Vp + ((size_t)(b * 64 + ct * 16 + l) << 12) + m0 + hs * 16;
      __builtin_prefetch(vr + 32, 0, 1);           // next V chunk
      const v16bf va = cat8(*(const v8bf*)vr, *(const v8bf*)(vr + 8));
      acc[ct] = WMMA_BF16(va, pb, acc[ct]);
    }
    asm volatile("s_wait_dscnt 0" ::: "memory");   // LDS reads done before overwrite
  }

  // Finalize: out matrix (c,n)-major is flat-identical to (w,h,c) output.
  const float g = gamma[0];
  #pragma unroll
  for (int ct = 0; ct < 4; ++ct) {
    #pragma unroll
    for (int j = 0; j < 8; ++j) {
      const int c = ct * 16 + hs * 8 + j;
      const size_t idx = ((size_t)b << 18) + ((size_t)c << 12) + n0 + l;
      out[idx] = g * acc[ct][j] + x[idx];
    }
  }
}

// ---------------------------------------------------------------------------
extern "C" void kernel_launch(void* const* d_in, const int* in_sizes, int n_in,
                              void* d_out, int out_size, void* d_ws, size_t ws_size,
                              hipStream_t stream) {
  const float* x  = (const float*)d_in[0];
  const float* Wq = (const float*)d_in[1];
  const float* bq = (const float*)d_in[2];
  const float* Wk = (const float*)d_in[3];
  const float* bk = (const float*)d_in[4];
  const float* Wv = (const float*)d_in[5];
  const float* bv = (const float*)d_in[6];
  const float* gm = (const float*)d_in[7];
  // d_in[8] = batch_size (shapes are compile-time here)

  float* out  = (float*)d_out;
  float* attn = out + (size_t)NB * 64 * 64 * 64;   // attention after out, flat

  // Workspace layout (bf16): Qp 1MB | Kp 1MB | Vp 2MB
  __bf16* Qp = (__bf16*)d_ws;
  __bf16* Kp = Qp + (size_t)NB * NN * 32;
  __bf16* Vp = Kp + (size_t)NB * NN * 32;

  proj_kernel<<<256, 128, 0, stream>>>(x, Wq, bq, Wk, bk, Wv, bv, Qp, Kp, Vp);
  attn_kernel<<<256, 128, 0, stream>>>(Qp, Kp, Vp, x, gm, out, attn);
}
